// SampleNet_69612829933915
// MI455X (gfx1250) — matrix operand order repaired and checked by reading, not compiled, not run on hardware
//
#include <hip/hip_runtime.h>

typedef __attribute__((ext_vector_type(16))) _Float16 v16h;
typedef __attribute__((ext_vector_type(8)))  float    v8f;

#define HH     112
#define WW     112
#define HWSZ   (112*112)
#define NB     8
#define NL     64
#define NM     16
#define NFR    (NB*NL)      // 512 frames
#define CIN    3
#define COUT   8
#define KKTOT  147          // 3*7*7
#define KPADT  160          // 5 K-blocks of 32
#define PD_EPS 1e-6f
#define BN_EPS 1e-5f
#define PATCH_W 120         // 118 used, padded
#define KSTRIDE 168         // a2col row stride in halves: 16B aligned, 21 quad-banks (odd) -> conflict-free
#define BSTRIDE 24          // bpack row stride in halves: 16B aligned, 3 quad-banks (odd)
#define PGRP   7            // frame pairs per pairdist block (63 = 9*7)

// ---------------------------------------------------------------------------
// K1: 7x7 conv (pad 3) as implicit GEMM via v_wmma_f32_16x16x32_f16.
// One workgroup per (row y, frame n); 7 waves, each owns a 16-pixel segment.
// im2col panel + B panel staged in LDS in fragment order: inner loop is pure
// ds_load_b128 + wmma with immediate offsets. Features stored f16.
// ---------------------------------------------------------------------------
__global__ __launch_bounds__(224)
void conv7x7_wmma(const float* __restrict__ x, const float* __restrict__ w,
                  const float* __restrict__ bias, _Float16* __restrict__ feat)
{
    const int y    = blockIdx.x;          // 0..111
    const int n    = blockIdx.y;          // 0..511
    const int tid  = threadIdx.x;
    const int lane = tid & 31;
    const int wave = tid >> 5;            // 0..6
    const int hi   = lane >> 4;           // half-wave
    const int r    = lane & 15;           // A row (pixel in tile)
    const int col  = lane & 15;           // B/D column (out channel)

    __shared__ float    patch[22 * PATCH_W];      // rows 0..20 = (c,fy); row 21 = zeros
    __shared__ float    wlds[COUT * KKTOT];
    __shared__ float    blds[COUT];
    __shared__ int      offtab[KPADT];            // K -> patch offset (zero row for K>=147)
    __shared__ _Float16 a2col[112 * KSTRIDE];     // [pixel][K] row-major, f16
    __shared__ _Float16 bpack[5 * 32 * BSTRIDE];  // [kb][lane][e] fragment-ordered, f16

    // ---- phase 1: stage weights, input patch, K-offset table
    for (int t = tid; t < COUT * KKTOT; t += 224) wlds[t] = w[t];
    if (tid < COUT) blds[tid] = bias[tid];
    if (tid < KPADT) {
        int K = tid, o;
        if (K < KKTOT) {
            int c  = K / 49;
            int rem = K - c * 49;
            int fy = rem / 7;
            int fx = rem - fy * 7;
            o = (c * 7 + fy) * PATCH_W + fx;
        } else {
            o = 21 * PATCH_W;                     // zero row
        }
        offtab[tid] = o;
    }
    for (int t = tid; t < 22 * 118; t += 224) {
        int row = t / 118;
        int cc  = t - row * 118;
        float v = 0.f;
        if (row < 21) {
            int c  = row / 7;
            int fy = row - c * 7;
            int yy = y + fy - 3;
            int xx = cc - 3;
            if ((unsigned)yy < 112u && (unsigned)xx < 112u)
                v = x[((size_t)(n * CIN + c)) * HWSZ + yy * WW + xx];
        }
        patch[row * PATCH_W + cc] = v;
    }
    __syncthreads();

    // ---- phase 2: build f16 im2col panel (K pairs -> packed u32 stores) + B panel
    for (int t = tid; t < (KPADT / 2) * 112; t += 224) {
        int kp = t / 112;                 // K-pair 0..79
        int rr = t - kp * 112;            // pixel 0..111
        int k0 = kp * 2;
        float v0 = patch[offtab[k0]     + rr];
        float v1 = patch[offtab[k0 + 1] + rr];
        union { _Float16 h[2]; unsigned u; } pk;
        pk.h[0] = (_Float16)v0;
        pk.h[1] = (_Float16)v1;
        *reinterpret_cast<unsigned*>(&a2col[rr * KSTRIDE + k0]) = pk.u;
    }
    for (int t = tid; t < 5 * 32 * 16; t += 224) {
        int kb = t >> 9;
        int ln = (t >> 4) & 31;
        int e  = t & 15;
        int K  = kb * 32 + (e >> 3) * 16 + (ln >> 4) * 8 + (e & 7);    // B layout: elem e <-> K = 16*(e/8) + 8*hi + e%8
        int cc = ln & 15;
        _Float16 v = (_Float16)0.f;
        if (cc < COUT && K < KKTOT) v = (_Float16)wlds[cc * KKTOT + K];
        bpack[(kb * 32 + ln) * BSTRIDE + e] = v;
    }
    __syncthreads();

    // ---- phase 3: 5 chained WMMAs per wave; fragment loads are b128 w/ imm offsets
    const int x0 = wave * 16;
    const _Float16* pa = &a2col[(x0 + r) * KSTRIDE + hi * 8];  // A: elems 0-7 @ K=kb*32+8*hi
    const _Float16* pb = &bpack[lane * BSTRIDE];
    v8f acc = {};
#pragma unroll
    for (int kb = 0; kb < 5; ++kb) {
        union { v16h v; uint4 q[2]; } A, B;
        A.q[0] = *reinterpret_cast<const uint4*>(pa + kb * 32);        // K=kb*32+8*hi ..+7
        A.q[1] = *reinterpret_cast<const uint4*>(pa + kb * 32 + 16);   // K=kb*32+16+8*hi ..+7
        B.q[0] = *reinterpret_cast<const uint4*>(pb + kb * 32 * BSTRIDE);
        B.q[1] = *reinterpret_cast<const uint4*>(pb + kb * 32 * BSTRIDE + 8);
        acc = __builtin_amdgcn_wmma_f32_16x16x32_f16(false, A.v, false, B.v,
                                                     (short)0, acc, false, false);
    }

    // D layout: VGPR v, lane l -> pixel M = v + 8*hi, channel N = l&15.
    if (col < COUT) {
        float b = blds[col];
        int px  = x0 + 8 * hi;
        _Float16* dst = feat + ((size_t)(n * COUT + col)) * HWSZ + y * WW + px;
        union { _Float16 h[8]; uint4 u; } pk;
#pragma unroll
        for (int v = 0; v < 8; ++v) pk.h[v] = (_Float16)(acc[v] + b);
        *reinterpret_cast<uint4*>(dst) = pk.u;   // 16B aligned: px % 8 == 0
    }
}

// ---------------------------------------------------------------------------
// K2: per-pixel L2 distance across 8 channels between consecutive frames,
// summed over H*W. One block per (b, pair-group of 7): walks frames forward
// keeping the previous frame's pixels in registers -> 8 frame-reads per 7
// pairs instead of 14. 32-bit loads = 2 pixels per VMEM op.
// ---------------------------------------------------------------------------
__global__ __launch_bounds__(256)
void pairdist(const _Float16* __restrict__ feat, float* __restrict__ dscore)
{
    const int pg  = blockIdx.x;           // 0..8
    const int b   = blockIdx.y;           // 0..7
    const int l0  = pg * PGRP;
    const int tid = threadIdx.x;
    const _Float16* base = feat + ((size_t)(b * NL + l0)) * COUT * HWSZ;

    float acc[PGRP];
#pragma unroll
    for (int g = 0; g < PGRP; ++g) acc[g] = 0.f;

    union U2 { unsigned u; _Float16 h[2]; };
    for (int j = tid; j < HWSZ / 2; j += 256) {
        U2 prev[COUT];
#pragma unroll
        for (int c = 0; c < COUT; ++c)
            prev[c].u = *reinterpret_cast<const unsigned*>(base + c * HWSZ + 2 * j);
        const _Float16* fp = base + (size_t)COUT * HWSZ;
#pragma unroll
        for (int g = 0; g < PGRP; ++g) {
            U2 cur[COUT];
#pragma unroll
            for (int c = 0; c < COUT; ++c)
                cur[c].u = *reinterpret_cast<const unsigned*>(fp + c * HWSZ + 2 * j);
            float s0 = 0.f, s1 = 0.f;
#pragma unroll
            for (int c = 0; c < COUT; ++c) {
                float d0 = (float)prev[c].h[0] - (float)cur[c].h[0] + PD_EPS;
                float d1 = (float)prev[c].h[1] - (float)cur[c].h[1] + PD_EPS;
                s0 += d0 * d0;
                s1 += d1 * d1;
            }
            acc[g] += sqrtf(s0) + sqrtf(s1);
#pragma unroll
            for (int c = 0; c < COUT; ++c) prev[c] = cur[c];
            fp += (size_t)COUT * HWSZ;
        }
    }

    __shared__ float red[256];
#pragma unroll
    for (int g = 0; g < PGRP; ++g) {
        red[tid] = acc[g];
        __syncthreads();
        for (int s = 128; s > 0; s >>= 1) {
            if (tid < s) red[tid] += red[tid + s];
            __syncthreads();
        }
        if (tid == 0) dscore[b * (NL - 1) + l0 + g] = red[0];
        __syncthreads();
    }
}

// ---------------------------------------------------------------------------
// K3: p = ds^sqrt(M/L) = sqrt(ds); normalize, cumsum, nearest-quantile argmin
// (first-min ties). One thread per batch element.
// ---------------------------------------------------------------------------
__global__ void select_frames(const float* __restrict__ dscore, int* __restrict__ idx)
{
    int b = threadIdx.x;
    if (b >= NB) return;
    float p[NL - 1];
    float sum = 0.f;
    for (int l = 0; l < NL - 1; ++l) {
        p[l] = sqrtf(dscore[b * (NL - 1) + l]);   // exponent = sqrt(16/64) = 0.5
        sum += p[l];
    }
    float inv = 1.f / sum;
    float cum = 0.f;
    float cums[NL - 1];
    for (int l = 0; l < NL - 1; ++l) { cum += p[l] * inv; cums[l] = cum; }
    for (int m = 0; m < NM; ++m) {
        float target = (float)m / (float)(NM - 1);
        float best = fabsf(cums[0] - target);
        int bi = 0;
        for (int l = 1; l < NL - 1; ++l) {
            float d = fabsf(cums[l] - target);
            if (d < best) { best = d; bi = l; }
        }
        idx[b * NM + m] = bi;
    }
}

__global__ void zero_stats(float* __restrict__ stats)
{
    if (threadIdx.x < 8) stats[threadIdx.x] = 0.f;
}

// ---------------------------------------------------------------------------
// K4a: 1x1 conv 8->3 on selected frames, accumulate per-channel sum/sumsq
// (training-mode batch stats). One block per (b, m). 2 pixels per VMEM op.
// ---------------------------------------------------------------------------
__global__ __launch_bounds__(256)
void bneck_stats(const _Float16* __restrict__ feat, const int* __restrict__ idx,
                 const float* __restrict__ w2, float* __restrict__ stats)
{
    const int m = blockIdx.x, b = blockIdx.y;
    const int fr = b * NL + idx[b * NM + m];
    const int tid = threadIdx.x;
    float wv[3][8];
#pragma unroll
    for (int o = 0; o < 3; ++o)
#pragma unroll
        for (int c = 0; c < 8; ++c) wv[o][c] = w2[o * 8 + c];
    const _Float16* f = feat + (size_t)fr * COUT * HWSZ;
    float s[3] = {0, 0, 0}, sq[3] = {0, 0, 0};
    union U2 { unsigned u; _Float16 h[2]; };
    for (int j = tid; j < HWSZ / 2; j += 256) {
        U2 fc[COUT];
#pragma unroll
        for (int c = 0; c < COUT; ++c)
            fc[c].u = *reinterpret_cast<const unsigned*>(f + c * HWSZ + 2 * j);
#pragma unroll
        for (int o = 0; o < 3; ++o) {
            float y0 = 0.f, y1 = 0.f;
#pragma unroll
            for (int c = 0; c < COUT; ++c) {
                y0 += wv[o][c] * (float)fc[c].h[0];
                y1 += wv[o][c] * (float)fc[c].h[1];
            }
            s[o]  += y0 + y1;
            sq[o] += y0 * y0 + y1 * y1;
        }
    }
    __shared__ float red[256];
    for (int jj = 0; jj < 6; ++jj) {
        red[tid] = (jj < 3) ? s[jj] : sq[jj - 3];
        __syncthreads();
        for (int st = 128; st > 0; st >>= 1) {
            if (tid < st) red[tid] += red[tid + st];
            __syncthreads();
        }
        if (tid == 0) atomicAdd(&stats[jj], red[0]);
        __syncthreads();
    }
}

// ---------------------------------------------------------------------------
// K4b: recompute y, normalize (gamma/beta), ReLU, add selected input frames.
// ---------------------------------------------------------------------------
__global__ __launch_bounds__(256)
void bneck_apply(const float* __restrict__ x, const _Float16* __restrict__ feat,
                 const int* __restrict__ idx, const float* __restrict__ w2,
                 const float* __restrict__ gamma, const float* __restrict__ beta,
                 const float* __restrict__ stats, float* __restrict__ out)
{
    const int m = blockIdx.x, b = blockIdx.y;
    const int fr = b * NL + idx[b * NM + m];
    const int tid = threadIdx.x;
    const float Ninv = 1.f / (float)(NB * NM * HWSZ);
    float scale[3], shift[3], wv[3][8];
#pragma unroll
    for (int o = 0; o < 3; ++o) {
        float mean = stats[o] * Ninv;
        float var  = stats[o + 3] * Ninv - mean * mean;
        float sc   = gamma[o] * rsqrtf(var + BN_EPS);
        scale[o] = sc;
        shift[o] = beta[o] - mean * sc;
#pragma unroll
        for (int c = 0; c < 8; ++c) wv[o][c] = w2[o * 8 + c];
    }
    const _Float16* f = feat + (size_t)fr * COUT * HWSZ;
    const float* xf = x + (size_t)fr * 3 * HWSZ;
    float* op = out + ((size_t)(b * NM + m)) * 3 * HWSZ;
    union U2 { unsigned u; _Float16 h[2]; };
    for (int j = tid; j < HWSZ / 2; j += 256) {
        U2 fc[COUT];
#pragma unroll
        for (int c = 0; c < COUT; ++c)
            fc[c].u = *reinterpret_cast<const unsigned*>(f + c * HWSZ + 2 * j);
#pragma unroll
        for (int o = 0; o < 3; ++o) {
            float y0 = 0.f, y1 = 0.f;
#pragma unroll
            for (int c = 0; c < COUT; ++c) {
                y0 += wv[o][c] * (float)fc[c].h[0];
                y1 += wv[o][c] * (float)fc[c].h[1];
            }
            y0 = fmaxf(y0 * scale[o] + shift[o], 0.f);
            y1 = fmaxf(y1 * scale[o] + shift[o], 0.f);
            float2 xv = *reinterpret_cast<const float2*>(xf + o * HWSZ + 2 * j);
            float2 ov = make_float2(xv.x + y0, xv.y + y1);
            *reinterpret_cast<float2*>(op + o * HWSZ + 2 * j) = ov;
        }
    }
}

// ---------------------------------------------------------------------------
extern "C" void kernel_launch(void* const* d_in, const int* in_sizes, int n_in,
                              void* d_out, int out_size, void* d_ws, size_t ws_size,
                              hipStream_t stream)
{
    const float* x       = (const float*)d_in[0];   // [8,64,3,112,112]
    const float* conv_w  = (const float*)d_in[1];   // [8,3,7,7]
    const float* conv_b  = (const float*)d_in[2];   // [8]
    const float* bneck_w = (const float*)d_in[3];   // [3,8,1,1]
    const float* gamma   = (const float*)d_in[4];   // [3]
    const float* beta    = (const float*)d_in[5];   // [3]

    // Workspace layout: f16 features (102.8 MB) + scores + stats + indices.
    char* ws = (char*)d_ws;
    _Float16* feat  = (_Float16*)ws;
    size_t featBytes = (size_t)NFR * COUT * HWSZ * 2;           // 102,760,448 B
    float* dscore = (float*)(ws + featBytes);                   // 504 floats
    float* stats  = dscore + NB * (NL - 1);                     // 8 floats (6 used)
    int*   idx    = (int*)(stats + 8);                          // 128 ints

    conv7x7_wmma <<<dim3(HH, NFR), 224, 0, stream>>>(x, conv_w, conv_b, feat);
    pairdist     <<<dim3((NL - 1) / PGRP, NB), 256, 0, stream>>>(feat, dscore);
    select_frames<<<1, 64, 0, stream>>>(dscore, idx);
    zero_stats   <<<1, 32, 0, stream>>>(stats);
    bneck_stats  <<<dim3(NM, NB), 256, 0, stream>>>(feat, idx, bneck_w, stats);
    bneck_apply  <<<dim3(NM, NB), 256, 0, stream>>>(x, feat, idx, bneck_w,
                                                    gamma, beta, stats, (float*)d_out);
}
